// SpanMaxPooler_32152125178212
// MI455X (gfx1250) — matrix operand order repaired and checked by reading, not compiled, not run on hardware
//
#include <hip/hip_runtime.h>
#include <hip/hip_bf16.h>
#include <math.h>

// ---------------------------------------------------------------------------
// Problem constants (match setup_inputs in the reference)
// ---------------------------------------------------------------------------
#define BB      16
#define SS      1024
#define HH      768
#define TT      64
#define MAXSPAN 4
#define LEDIM   32
#define DIN     (3 * HH + LEDIM)   // 2336
#define NMAX    (SS * MAXSPAN)     // 4096 (>= true span count N)
#define MTOT    (BB * NMAX)        // 65536 GEMM rows

typedef unsigned short u16;
typedef unsigned int   u32;
typedef __bf16 bf16_t;
typedef bf16_t v16bf __attribute__((ext_vector_type(16)));
typedef float  v8f   __attribute__((ext_vector_type(8)));
typedef u32    v4u   __attribute__((ext_vector_type(4)));

union FragBF { v4u q[2]; v16bf v; };   // 32B: one 16x32 bf16 A-frag or 32x16 B-frag

__device__ __forceinline__ u16 f2bf(float f) {
  u32 u = __float_as_uint(f);
  u32 r = u + 0x7fffu + ((u >> 16) & 1u);   // round-to-nearest-even
  return (u16)(r >> 16);
}
// Branch-free erf (Abramowitz-Stegun 7.1.26, |err| < 1.5e-7): v_rcp + v_exp,
// no EXEC divergence (libm erff inlines a large branchy poly).
__device__ __forceinline__ float erf_fast(float x) {
  float ax = fabsf(x);
  float t  = __builtin_amdgcn_rcpf(1.0f + 0.3275911f * ax);
  float p  = t * (0.254829592f +
             t * (-0.284496736f +
             t * (1.421413741f +
             t * (-1.453152027f +
             t * 1.061405429f))));
  float r = 1.0f - p * __expf(-ax * ax);
  return copysignf(r, x);
}
__device__ __forceinline__ float geluf(float x) {
  return 0.5f * x * (1.0f + erf_fast(x * 0.70710678118654752f));
}
// order-preserving float <-> uint encoding for atomicMax
__device__ __forceinline__ u32 encF(float f) {
  u32 u = __float_as_uint(f);
  return (u >> 31) ? ~u : (u | 0x80000000u);
}
__device__ __forceinline__ float decF(u32 k) {
  return (k & 0x80000000u) ? __uint_as_float(k ^ 0x80000000u) : __uint_as_float(~k);
}

// ---------------------------------------------------------------------------
// Span enumeration: off(i) = sum_{k<i} min(L, n-k), closed form
// ---------------------------------------------------------------------------
__device__ __forceinline__ int span_offset(int i, int n) {
  int t1 = n - MAXSPAN + 1; if (t1 < 0) t1 = 0;
  if (i <= t1) return i * MAXSPAN;
  int dmax = n - t1;
  int a = n - i + 1;
  return t1 * MAXSPAN + (dmax * (dmax + 1) - (a - 1) * a) / 2;
}

__global__ __launch_bounds__(1024) void k_build_spans(
    const unsigned char* __restrict__ text_mask, int* __restrict__ meta,
    int* __restrict__ starts, int* __restrict__ ends, int* __restrict__ lens) {
  __shared__ int scan[SS];
  __shared__ int pos[SS];
  const int t = threadIdx.x;
  const int m = text_mask[t] ? 1 : 0;
  scan[t] = m;
  __syncthreads();
  for (int off = 1; off < SS; off <<= 1) {   // Hillis-Steele inclusive scan
    int v = scan[t];
    int add = (t >= off) ? scan[t - off] : 0;
    __syncthreads();
    scan[t] = v + add;
    __syncthreads();
  }
  const int n = scan[SS - 1];
  if (m) pos[scan[t] - 1] = t;
  __syncthreads();
  if (t < n) {
    int cnt = n - t; if (cnt > MAXSPAN) cnt = MAXSPAN;
    int off = span_offset(t, n);
    for (int c = 0; c < cnt; ++c) {
      starts[off + c] = pos[t];
      ends[off + c]   = pos[t + c];
      lens[off + c]   = c + 1;
    }
  }
  const int N = span_offset(n, n);
  if (t == 0) { meta[0] = N; meta[1] = n; }
  for (int s = N + t; s < NMAX; s += 1024) { starts[s] = 0; ends[s] = 0; lens[s] = 1; }
}

// ---------------------------------------------------------------------------
// f32 -> bf16 bulk convert
// ---------------------------------------------------------------------------
__global__ __launch_bounds__(256) void k_f32_to_bf16(
    const float* __restrict__ src, u16* __restrict__ dst, int count) {
  int i = blockIdx.x * 256 + threadIdx.x;
  if (i < count) dst[i] = f2bf(src[i]);
}

// ---------------------------------------------------------------------------
// Feature rows: [start | end | mean | len_emb] -> bf16, row stride DIN
// ---------------------------------------------------------------------------
__global__ __launch_bounds__(256) void k_build_feat(
    const float* __restrict__ hid, const float* __restrict__ len_emb,
    const int* __restrict__ starts, const int* __restrict__ ends,
    const int* __restrict__ lens, u16* __restrict__ Fb) {
  const int idx = blockIdx.x;              // b * NMAX + s
  const int s = idx & (NMAX - 1);
  const int b = idx >> 12;
  const int st = starts[s], en = ends[s], ln = lens[s];
  const float* hb = hid + (size_t)b * SS * HH;
  const float inv = 1.0f / (float)ln;
  const size_t base = (size_t)idx * DIN;
  for (int k = threadIdx.x; k < DIN; k += 256) {
    float v;
    if (k < HH) {
      v = hb[(size_t)st * HH + k];
    } else if (k < 2 * HH) {
      v = hb[(size_t)en * HH + (k - HH)];
    } else if (k < 3 * HH) {
      int kk = k - 2 * HH;
      float acc = 0.0f;
      for (int r = st; r <= en; ++r) acc += hb[(size_t)r * HH + kk];
      v = acc * inv;
    } else {
      v = len_emb[ln * LEDIM + (k - 3 * HH)];
    }
    Fb[base + k] = f2bf(v);
  }
}

// ---------------------------------------------------------------------------
// bf16 WMMA GEMM: D = act(A(MxK) * Bm(KxN) + bias)
//   128x64 block tile, 8 waves; wave tile = 32M x 32N -> 4 wmma / k-step.
//   Double-buffered LDS staging: one barrier per k-step, global loads for
//   tile k+1 overlap WMMA on tile k.
//   act==1: GELU -> bf16 out (Dbf);  act==0: f32 out (Df)
// ---------------------------------------------------------------------------
#define LDSTRIDE 40   // bf16 elements per LDS row (80B, 16B multiple)
#define BMT 128
#define BNT 64

__global__ __launch_bounds__(256) void k_gemm_bf16(
    const u16* __restrict__ A, const u16* __restrict__ Bm,
    const float* __restrict__ bias, u16* __restrict__ Dbf,
    float* __restrict__ Df, int K, int Nc, int act) {
  __shared__ u16 Asm[2][BMT * LDSTRIDE];   // 2 x 10240 B
  __shared__ u16 Bsm[2][BNT * LDSTRIDE];   // 2 x  5120 B
  const int tid  = threadIdx.x;
  const int lane = tid & 31;
  const int wave = tid >> 5;
  const int wm = wave & 3;        // 32-row chunk within 128-row tile
  const int wn = wave >> 2;       // 32-col half within 64-col tile
  const size_t m0 = (size_t)blockIdx.x * BMT;
  const int n0 = blockIdx.y * BNT;

  v8f acc[2][2];
  #pragma unroll
  for (int i = 0; i < 2; ++i)
    #pragma unroll
    for (int j = 0; j < 2; ++j)
      #pragma unroll
      for (int r = 0; r < 8; ++r) acc[i][j][r] = 0.0f;

  const int ar = tid >> 1, ac = (tid & 1) * 16;  // A-stage: 128 rows x 32 k
  const int bk = tid >> 3, bn = (tid & 7) * 8;   // B-stage: 32 k x 64 n

  const int mr0 = wm * 32 + (lane & 15);
  const int g   = lane >> 4;
  const int nc0 = wn * 32 + (lane & 15);
  const int nc1 = nc0 + 16;

  const u16* aptr = A + (m0 + ar) * K + ac;
  const u16* bptr = Bm + (size_t)bk * Nc + n0 + bn;

  auto stage = [&](int buf, int k0) {
    v4u av0 = *reinterpret_cast<const v4u*>(aptr + k0);
    v4u av1 = *reinterpret_cast<const v4u*>(aptr + k0 + 8);
    *reinterpret_cast<v4u*>(&Asm[buf][ar * LDSTRIDE + ac])     = av0;
    *reinterpret_cast<v4u*>(&Asm[buf][ar * LDSTRIDE + ac + 8]) = av1;
    v4u bv = *reinterpret_cast<const v4u*>(bptr + (size_t)k0 * Nc);
    const u16* bs = reinterpret_cast<const u16*>(&bv);
    #pragma unroll
    for (int i = 0; i < 8; ++i) Bsm[buf][(bn + i) * LDSTRIDE + bk] = bs[i];
  };

  const int KT = K >> 5;
  stage(0, 0);
  for (int kt = 0; kt < KT; ++kt) {
    const int cur = kt & 1;
    __syncthreads();
    if (kt + 1 < KT) {
      stage(cur ^ 1, (kt + 1) << 5);
      __builtin_prefetch(aptr + ((kt + 2) << 5), 0, 3);               // near-cache hint
      __builtin_prefetch(bptr + ((size_t)((kt + 2) << 5)) * Nc, 0, 3);
    }
    // A frags: lane holds M=lane%16; VGPR pairs cover K per 16-bit A layout
    FragBF a0, a1, b0, b1;
    const u16* as = &Asm[cur][0];
    const u16* bsrc = &Bsm[cur][0];
    a0.q[0] = *reinterpret_cast<const v4u*>(as + mr0 * LDSTRIDE + 8 * g);
    a0.q[1] = *reinterpret_cast<const v4u*>(as + mr0 * LDSTRIDE + 16 + 8 * g);
    a1.q[0] = *reinterpret_cast<const v4u*>(as + (mr0 + 16) * LDSTRIDE + 8 * g);
    a1.q[1] = *reinterpret_cast<const v4u*>(as + (mr0 + 16) * LDSTRIDE + 16 + 8 * g);
    // B frags: lane holds N=lane%16; K = 8*g + e and 16 + 8*g + e (same order as A) in transposed LDS
    b0.q[0] = *reinterpret_cast<const v4u*>(bsrc + nc0 * LDSTRIDE + 8 * g);
    b0.q[1] = *reinterpret_cast<const v4u*>(bsrc + nc0 * LDSTRIDE + 16 + 8 * g);
    b1.q[0] = *reinterpret_cast<const v4u*>(bsrc + nc1 * LDSTRIDE + 8 * g);
    b1.q[1] = *reinterpret_cast<const v4u*>(bsrc + nc1 * LDSTRIDE + 16 + 8 * g);

    acc[0][0] = __builtin_amdgcn_wmma_f32_16x16x32_bf16(false, a0.v, false, b0.v,
                                                        (short)0, acc[0][0], false, false);
    acc[0][1] = __builtin_amdgcn_wmma_f32_16x16x32_bf16(false, a0.v, false, b1.v,
                                                        (short)0, acc[0][1], false, false);
    acc[1][0] = __builtin_amdgcn_wmma_f32_16x16x32_bf16(false, a1.v, false, b0.v,
                                                        (short)0, acc[1][0], false, false);
    acc[1][1] = __builtin_amdgcn_wmma_f32_16x16x32_bf16(false, a1.v, false, b1.v,
                                                        (short)0, acc[1][1], false, false);
  }

  // epilogue: C/D layout -> row = +r + 8*g, col = lane%16 within each 16x16 tile
  const int cols[2] = { n0 + nc0, n0 + nc1 };
  const float bb[2] = { bias[cols[0]], bias[cols[1]] };
  #pragma unroll
  for (int i = 0; i < 2; ++i) {
    #pragma unroll
    for (int j = 0; j < 2; ++j) {
      #pragma unroll
      for (int r = 0; r < 8; ++r) {
        size_t row = m0 + wm * 32 + i * 16 + r + 8 * g;
        float v = acc[i][j][r] + bb[j];
        if (act) Dbf[row * Nc + cols[j]] = f2bf(geluf(v));
        else     Df[row * Nc + cols[j]]  = v;
      }
    }
  }
}

// ---------------------------------------------------------------------------
// Row L2-normalize (f32 in) -> bf16 out; one block per row of width HH
// ---------------------------------------------------------------------------
__global__ __launch_bounds__(256) void k_rownorm(
    const float* __restrict__ src, u16* __restrict__ dst) {
  __shared__ float red[256];
  const size_t row = blockIdx.x;
  const float* x = src + row * HH;
  float p = 0.0f;
  for (int k = threadIdx.x; k < HH; k += 256) { float v = x[k]; p += v * v; }
  red[threadIdx.x] = p;
  __syncthreads();
  for (int s = 128; s > 0; s >>= 1) {
    if (threadIdx.x < s) red[threadIdx.x] += red[threadIdx.x + s];
    __syncthreads();
  }
  const float sc = 1.0f / fmaxf(sqrtf(red[0]), 1e-12f);
  u16* d = dst + row * HH;
  for (int k = threadIdx.x; k < HH; k += 256) d[k] = f2bf(x[k] * sc);
}

// ---------------------------------------------------------------------------
// sim = topic_vec (64xH) x span_vec^T (Hx64-tile) via WMMA, fused
// topic-mask + max over topics -> span_scores[b, n0..n0+63]
// ---------------------------------------------------------------------------
__global__ __launch_bounds__(256) void k_sim_max(
    const u16* __restrict__ TV, const u16* __restrict__ SV,
    const unsigned char* __restrict__ topic_mask, float* __restrict__ scores) {
  __shared__ u16 Asm[64 * LDSTRIDE];
  __shared__ u16 Bsm[64 * LDSTRIDE];
  __shared__ float red[4][64];
  __shared__ unsigned char tmk[TT];
  const int tid  = threadIdx.x;
  const int lane = tid & 31;
  const int wave = tid >> 5;
  const int wm = wave & 3;
  const int wn = wave >> 2;
  const int b  = blockIdx.y;
  const int n0 = blockIdx.x * 64;
  if (tid < TT) tmk[tid] = topic_mask[b * TT + tid];

  v8f acc0, acc1;
  #pragma unroll
  for (int i = 0; i < 8; ++i) { acc0[i] = 0.0f; acc1[i] = 0.0f; }

  const int ar = tid >> 2, ac = (tid & 3) * 8;   // 64 rows x 32 k stage
  const int mr = wm * 16 + (lane & 15);
  const int g  = lane >> 4;
  const int nc0 = wn * 32 + (lane & 15);
  const int nc1 = nc0 + 16;
  const u16* Ab = TV + (size_t)b * TT * HH;                 // topics, row-major K
  const u16* Bb = SV + ((size_t)b * NMAX + n0) * HH;        // span rows = B columns

  for (int k0 = 0; k0 < HH; k0 += 32) {
    v4u av = *reinterpret_cast<const v4u*>(Ab + (size_t)ar * HH + k0 + ac);
    *reinterpret_cast<v4u*>(&Asm[ar * LDSTRIDE + ac]) = av;
    v4u bv = *reinterpret_cast<const v4u*>(Bb + (size_t)ar * HH + k0 + ac);
    *reinterpret_cast<v4u*>(&Bsm[ar * LDSTRIDE + ac]) = bv;  // already n-major
    __syncthreads();

    FragBF afr, bf0, bf1;
    afr.q[0] = *reinterpret_cast<const v4u*>(&Asm[mr * LDSTRIDE + 8 * g]);
    afr.q[1] = *reinterpret_cast<const v4u*>(&Asm[mr * LDSTRIDE + 16 + 8 * g]);
    bf0.q[0] = *reinterpret_cast<const v4u*>(&Bsm[nc0 * LDSTRIDE + 8 * g]);
    bf0.q[1] = *reinterpret_cast<const v4u*>(&Bsm[nc0 * LDSTRIDE + 16 + 8 * g]);
    bf1.q[0] = *reinterpret_cast<const v4u*>(&Bsm[nc1 * LDSTRIDE + 8 * g]);
    bf1.q[1] = *reinterpret_cast<const v4u*>(&Bsm[nc1 * LDSTRIDE + 16 + 8 * g]);

    acc0 = __builtin_amdgcn_wmma_f32_16x16x32_bf16(false, afr.v, false, bf0.v,
                                                   (short)0, acc0, false, false);
    acc1 = __builtin_amdgcn_wmma_f32_16x16x32_bf16(false, afr.v, false, bf1.v,
                                                   (short)0, acc1, false, false);
    __syncthreads();
  }

  // masked max over this wave's 16 topic rows (r + 8*g layout)
  const float NEGINF = -__builtin_inff();
  float m0v = NEGINF, m1v = NEGINF;
  #pragma unroll
  for (int r = 0; r < 8; ++r) {
    int tpc = wm * 16 + r + 8 * g;
    if (tmk[tpc]) {
      m0v = fmaxf(m0v, acc0[r]);
      m1v = fmaxf(m1v, acc1[r]);
    }
  }
  m0v = fmaxf(m0v, __shfl_xor(m0v, 16));   // combine M=r with M=r+8 halves
  m1v = fmaxf(m1v, __shfl_xor(m1v, 16));
  if (lane < 16) {
    red[wm][wn * 32 + lane]      = m0v;
    red[wm][wn * 32 + 16 + lane] = m1v;
  }
  __syncthreads();
  if (tid < 64) {
    float m = fmaxf(fmaxf(red[0][tid], red[1][tid]),
                    fmaxf(red[2][tid], red[3][tid]));
    scores[(size_t)b * NMAX + n0 + tid] = m;
  }
}

// ---------------------------------------------------------------------------
// Token scoring: ordered-uint atomic max scatter, then decode + text mask
// ---------------------------------------------------------------------------
__global__ __launch_bounds__(256) void k_init_tok(u32* __restrict__ enc) {
  int i = blockIdx.x * 256 + threadIdx.x;
  if (i < BB * SS) enc[i] = 0x007FFFFFu;   // encF(-inf)
}

__global__ __launch_bounds__(256) void k_scatter(
    const float* __restrict__ scores, const int* __restrict__ starts,
    const int* __restrict__ ends, const int* __restrict__ meta,
    u32* __restrict__ enc) {
  const int idx = blockIdx.x * 256 + threadIdx.x;
  const int s = idx & (NMAX - 1);
  const int b = idx >> 12;
  if (s >= meta[0]) return;
  const u32 key = encF(scores[(size_t)b * NMAX + s]);
  u32* eb = enc + b * SS;
  const int st = starts[s], en = ends[s];
  for (int t = st; t <= en; ++t) atomicMax(&eb[t], key);
}

__global__ __launch_bounds__(256) void k_finalize(
    const u32* __restrict__ enc, const unsigned char* __restrict__ text_mask,
    float* __restrict__ out) {
  int i = blockIdx.x * 256 + threadIdx.x;
  if (i < BB * SS) out[i] = text_mask[i] ? decF(enc[i]) : -__builtin_inff();
}

// ---------------------------------------------------------------------------
// Launch
// ---------------------------------------------------------------------------
extern "C" void kernel_launch(void* const* d_in, const int* in_sizes, int n_in,
                              void* d_out, int out_size, void* d_ws, size_t ws_size,
                              hipStream_t stream) {
  (void)in_sizes; (void)n_in; (void)out_size; (void)ws_size;
  const float* hid        = (const float*)d_in[0];
  const float* topic      = (const float*)d_in[1];
  const unsigned char* tm = (const unsigned char*)d_in[2];  // topic_mask
  const unsigned char* xm = (const unsigned char*)d_in[3];  // text_mask
  const float* len_emb    = (const float*)d_in[4];
  const float* W1         = (const float*)d_in[5];
  const float* b1         = (const float*)d_in[6];
  const float* W2         = (const float*)d_in[7];
  const float* b2         = (const float*)d_in[8];
  float* out = (float*)d_out;

  char* ws = (char*)d_ws;
  size_t o = 0;
  auto alloc = [&](size_t bytes) { size_t r = o; o += (bytes + 255) & ~(size_t)255; return r; };
  const size_t o_meta   = alloc(2 * sizeof(int));
  const size_t o_starts = alloc((size_t)NMAX * 4);
  const size_t o_ends   = alloc((size_t)NMAX * 4);
  const size_t o_lens   = alloc((size_t)NMAX * 4);
  const size_t o_w1b    = alloc((size_t)DIN * HH * 2);
  const size_t o_w2b    = alloc((size_t)HH * HH * 2);
  const size_t o_tvb    = alloc((size_t)BB * TT * HH * 2);
  const size_t o_scores = alloc((size_t)MTOT * 4);
  const size_t o_enc    = alloc((size_t)BB * SS * 4);
  const size_t o_g      = alloc((size_t)MTOT * HH * 2);   // G (gelu out, bf16)
  const size_t o_fb     = alloc((size_t)MTOT * DIN * 2);  // feat bf16
  const size_t o_o2     = o_fb;   // GEMM2 f32 out aliases feat (feat dead)
  const size_t o_svb    = o_g;    // normalized span vecs alias G (G dead)

  int*   meta   = (int*)(ws + o_meta);
  int*   starts = (int*)(ws + o_starts);
  int*   ends   = (int*)(ws + o_ends);
  int*   lens   = (int*)(ws + o_lens);
  u16*   W1b    = (u16*)(ws + o_w1b);
  u16*   W2b    = (u16*)(ws + o_w2b);
  u16*   TVb    = (u16*)(ws + o_tvb);
  float* scores = (float*)(ws + o_scores);
  u32*   enc    = (u32*)(ws + o_enc);
  u16*   G      = (u16*)(ws + o_g);
  u16*   Fb     = (u16*)(ws + o_fb);
  float* O2     = (float*)(ws + o_o2);
  u16*   SVb    = (u16*)(ws + o_svb);

  // 1. spans
  k_build_spans<<<1, 1024, 0, stream>>>(xm, meta, starts, ends, lens);
  // 2. weight conversion
  {
    int c1 = DIN * HH, c2 = HH * HH;
    k_f32_to_bf16<<<(c1 + 255) / 256, 256, 0, stream>>>(W1, W1b, c1);
    k_f32_to_bf16<<<(c2 + 255) / 256, 256, 0, stream>>>(W2, W2b, c2);
  }
  // 3. features (bf16, row stride DIN)
  k_build_feat<<<MTOT, 256, 0, stream>>>(hid, len_emb, starts, ends, lens, Fb);
  // 4. GEMM1: G = gelu(Fb @ W1 + b1)   [M=65536, K=2336, N=768]
  k_gemm_bf16<<<dim3(MTOT / BMT, HH / BNT), 256, 0, stream>>>(
      Fb, W1b, b1, G, nullptr, DIN, HH, 1);
  // 5. GEMM2: O2 = G @ W2 + b2         [M=65536, K=768, N=768]
  k_gemm_bf16<<<dim3(MTOT / BMT, HH / BNT), 256, 0, stream>>>(
      G, W2b, b2, nullptr, O2, HH, HH, 0);
  // 6. normalize span vectors and topic vectors
  k_rownorm<<<MTOT, 256, 0, stream>>>(O2, SVb);
  k_rownorm<<<BB * TT, 256, 0, stream>>>(topic, TVb);
  // 7. sim + masked max over topics
  k_sim_max<<<dim3(NMAX / 64, BB), 256, 0, stream>>>(TVb, SVb, tm, scores);
  // 8. token max
  k_init_tok<<<(BB * SS + 255) / 256, 256, 0, stream>>>(enc);
  k_scatter<<<MTOT / 256, 256, 0, stream>>>(scores, starts, ends, meta, enc);
  k_finalize<<<(BB * SS + 255) / 256, 256, 0, stream>>>(enc, xm, out);
}
